// GAT_20151986552834
// MI455X (gfx1250) — matrix operand order repaired and checked by reading, not compiled, not run on hardware
//
#include <hip/hip_runtime.h>
#include <hip/hip_bf16.h>

typedef __attribute__((ext_vector_type(16))) __bf16 v16bf;
typedef __attribute__((ext_vector_type(8)))  __bf16 v8bf;
typedef __attribute__((ext_vector_type(8)))  float  v8f;

#define NN    10000
#define EE    160000
#define FIN   256
#define H1H   8
#define C1C   128
#define HC1   1024
#define C2C   128
#define NSLOPE 0.2f

// ---------- monotone float<->uint encoding for atomic max ----------
__device__ __forceinline__ unsigned fenc(float x) {
  unsigned u = __float_as_uint(x);
  return (u & 0x80000000u) ? ~u : (u | 0x80000000u);
}
__device__ __forceinline__ float fdec(unsigned u) {
  return __uint_as_float((u & 0x80000000u) ? (u & 0x7FFFFFFFu) : ~u);
}

// ---------- utility kernels ----------
__global__ void k_fill(unsigned* __restrict__ p, unsigned v, long n) {
  long i = blockIdx.x * (long)blockDim.x + threadIdx.x;
  long st = (long)gridDim.x * blockDim.x;
  for (; i < n; i += st) p[i] = v;
}

__global__ void k_cvt_bf16(const float* __restrict__ s, __bf16* __restrict__ d, long n) {
  long i = blockIdx.x * (long)blockDim.x + threadIdx.x;
  long st = (long)gridDim.x * blockDim.x;
  for (; i < n; i += st) d[i] = (__bf16)s[i];
}

// W[K,Nc] row-major -> Wt[Nc,K] bf16 (so WMMA B fragments load contiguously)
__global__ void k_transpose_bf16(const float* __restrict__ W, __bf16* __restrict__ Wt,
                                 int K, int Nc) {
  long i = blockIdx.x * (long)blockDim.x + threadIdx.x;
  if (i >= (long)K * Nc) return;
  int k = (int)(i / Nc), c = (int)(i % Nc);
  Wt[(long)c * K + k] = (__bf16)W[i];
}

// ---------- WMMA fragment loaders ----------
// A 16x32 bf16 fragment: elems 0..7 at p, 8..15 at p+16 (lane half pre-applied)
__device__ __forceinline__ v16bf ldA(const __bf16* p) {
  v8bf x0 = *(const v8bf*)(p);
  v8bf x1 = *(const v8bf*)(p + 16);
  return __builtin_shufflevector(x0, x1, 0,1,2,3,4,5,6,7,8,9,10,11,12,13,14,15);
}
// B 32x16 bf16 fragment: elems 0..7 at p, 8..15 at p+16 (lane half pre-applied)
__device__ __forceinline__ v16bf ldB(const __bf16* p) {
  v8bf x0 = *(const v8bf*)(p);
  v8bf x1 = *(const v8bf*)(p + 16);
  return __builtin_shufflevector(x0, x1, 0,1,2,3,4,5,6,7,8,9,10,11,12,13,14,15);
}

#define WMMA_BF16(a, b, c) \
  __builtin_amdgcn_wmma_f32_16x16x32_bf16(false, (a), false, (b), (short)0, (c), false, false)

// ---------- WMMA GEMM: C[M,Nn] = A[M,K] * Bt[Nn,K]^T, bf16 in / f32 out ----------
// One wave computes a 16x64 tile (4 accumulators). Fragments are double-buffered with
// plain named registers: the next K-step's loads are issued before the current step's
// 4 WMMAs so latency hides behind matrix work without spilling (~112 live VGPRs).
// Requires M%16==0, Nn%64==0, K%64==0 (holds for 10000/1024/256 and 10000/128/1024).
__global__ void k_gemm_bf16_wmma(const __bf16* __restrict__ A,
                                 const __bf16* __restrict__ Bt,
                                 float* __restrict__ C,
                                 int M, int Nn, int K) {
  const int lane = threadIdx.x & 31;
  const int wave = (int)((blockIdx.x * (long)blockDim.x + threadIdx.x) >> 5);
  const int tilesN = Nn >> 6;                 // 64-wide N tiles
  const int tilesM = M >> 4;                  // 16-high M tiles (M%16==0)
  if (wave >= tilesM * tilesN) return;
  const int tm = wave / tilesN;
  const int tn = wave % tilesN;
  const int half = lane >> 4;                 // 0: lanes 0-15, 1: lanes 16-31
  const int l15  = lane & 15;

  // pre-offset base pointers by the lane-half K sub-offsets
  const __bf16* Ap = A  + (long)(tm * 16 + l15) * K + half * 8;    // A: klo = half*8
  const __bf16* Bp = Bt + (long)(tn * 64 + l15) * K + half * 8;    // B: klo = half*8
  const long bs = 16L * K;

  v8f acc0 = {}, acc1 = {}, acc2 = {}, acc3 = {};

  // pipeline prologue: fragments for k=0
  v16bf a0  = ldA(Ap);
  v16bf b00 = ldB(Bp);
  v16bf b01 = ldB(Bp + bs);
  v16bf b02 = ldB(Bp + 2 * bs);
  v16bf b03 = ldB(Bp + 3 * bs);

  for (int k0 = 0; k0 < K; k0 += 64) {
    // issue loads for k0+32 while the k0 WMMAs run
    const __bf16* Ap1 = Ap + k0 + 32;
    const __bf16* Bp1 = Bp + k0 + 32;
    v16bf a1  = ldA(Ap1);
    v16bf b10 = ldB(Bp1);
    v16bf b11 = ldB(Bp1 + bs);
    v16bf b12 = ldB(Bp1 + 2 * bs);
    v16bf b13 = ldB(Bp1 + 3 * bs);

    acc0 = WMMA_BF16(a0, b00, acc0);
    acc1 = WMMA_BF16(a0, b01, acc1);
    acc2 = WMMA_BF16(a0, b02, acc2);
    acc3 = WMMA_BF16(a0, b03, acc3);

    if (k0 + 64 < K) {            // issue loads for k0+64 while the k0+32 WMMAs run
      const __bf16* Ap2 = Ap + k0 + 64;
      const __bf16* Bp2 = Bp + k0 + 64;
      a0  = ldA(Ap2);
      b00 = ldB(Bp2);
      b01 = ldB(Bp2 + bs);
      b02 = ldB(Bp2 + 2 * bs);
      b03 = ldB(Bp2 + 3 * bs);
    }

    acc0 = WMMA_BF16(a1, b10, acc0);
    acc1 = WMMA_BF16(a1, b11, acc1);
    acc2 = WMMA_BF16(a1, b12, acc2);
    acc3 = WMMA_BF16(a1, b13, acc3);
  }

  // C/D layout: VGPR v -> row tm*16 + half*8 + v, col tn*64 + t*16 + l15
  const int cb = tn * 64 + l15;
  const long rb = (long)(tm * 16 + half * 8) * Nn;
#pragma unroll
  for (int v = 0; v < 8; ++v) {
    C[rb + (long)v * Nn + cb]      = acc0[v];
    C[rb + (long)v * Nn + cb + 16] = acc1[v];
    C[rb + (long)v * Nn + cb + 32] = acc2[v];
    C[rb + (long)v * Nn + cb + 48] = acc3[v];
  }
}

// ---------- attention scores: wave per (node,head), coalesced + shfl reduction ----------
template <int HEADS, int C>
__global__ void k_scores(const float* __restrict__ H, const float* __restrict__ a_s,
                         const float* __restrict__ a_d, float* __restrict__ ss,
                         float* __restrict__ sd, int n) {
  const int lane = threadIdx.x & 31;
  const int i = (int)((blockIdx.x * (long)blockDim.x + threadIdx.x) >> 5);
  if (i >= n * HEADS) return;
  const int h = i % HEADS;
  const float* hp  = H + (long)i * C;          // H is [n, HEADS*C] row-major; i = node*HEADS+h
  const float* asp = a_s + (long)h * C;
  const float* adp = a_d + (long)h * C;
  float s0 = 0.f, s1 = 0.f;
#pragma unroll
  for (int c = lane; c < C; c += 32) {
    float v = hp[c];
    s0 += v * asp[c];
    s1 += v * adp[c];
  }
#pragma unroll
  for (int off = 16; off > 0; off >>= 1) {
    s0 += __shfl_xor(s0, off, 32);
    s1 += __shfl_xor(s1, off, 32);
  }
  if (lane == 0) { ss[i] = s0; sd[i] = s1; }
}

// ---------- edge pass 1: segment max (encoded-uint atomic max) ----------
template <int HEADS>
__global__ void k_edge_max(const int* __restrict__ ei, const float* __restrict__ ss,
                           const float* __restrict__ sd, unsigned* __restrict__ m) {
  long i = blockIdx.x * (long)blockDim.x + threadIdx.x;
  if (i >= (long)(EE + NN) * HEADS) return;
  int t = (int)(i / HEADS), h = (int)(i % HEADS);
  int s = (t < EE) ? ei[t]      : (t - EE);
  int d = (t < EE) ? ei[EE + t] : (t - EE);
  float e = ss[s * HEADS + h] + sd[d * HEADS + h];
  e = (e > 0.f) ? e : NSLOPE * e;
  atomicMax(m + (long)d * HEADS + h, fenc(e));
}

// ---------- edge pass 2: denom = sum exp(e - m[dst]) ----------
template <int HEADS>
__global__ void k_edge_sum(const int* __restrict__ ei, const float* __restrict__ ss,
                           const float* __restrict__ sd, const unsigned* __restrict__ m,
                           float* __restrict__ den) {
  long i = blockIdx.x * (long)blockDim.x + threadIdx.x;
  if (i >= (long)(EE + NN) * HEADS) return;
  int t = (int)(i / HEADS), h = (int)(i % HEADS);
  int s = (t < EE) ? ei[t]      : (t - EE);
  int d = (t < EE) ? ei[EE + t] : (t - EE);
  float e = ss[s * HEADS + h] + sd[d * HEADS + h];
  e = (e > 0.f) ? e : NSLOPE * e;
  float p = __expf(e - fdec(m[(long)d * HEADS + h]));
  atomicAdd(den + (long)d * HEADS + h, p);
}

// ---------- edge pass 3: out[dst] += alpha * H[src], one wave per edge ----------
template <int HEADS, int C>
__global__ void k_edge_agg(const int* __restrict__ ei, const float* __restrict__ ss,
                           const float* __restrict__ sd, const unsigned* __restrict__ m,
                           const float* __restrict__ den, const float* __restrict__ H,
                           float* __restrict__ out) {
  const int lane = threadIdx.x & 31;
  long wave = (blockIdx.x * (long)blockDim.x + threadIdx.x) >> 5;
  long nw   = ((long)gridDim.x * blockDim.x) >> 5;
  for (long t = wave; t < (long)(EE + NN); t += nw) {
    int s = (t < EE) ? ei[t]      : (int)(t - EE);
    int d = (t < EE) ? ei[EE + t] : (int)(t - EE);
    const float* hs = H + (long)s * HEADS * C;
    __builtin_prefetch(hs + lane, 0, 0);       // global_prefetch_b8 into caches
    float* od = out + (long)d * HEADS * C;
#pragma unroll
    for (int h = 0; h < HEADS; ++h) {
      float e = ss[s * HEADS + h] + sd[d * HEADS + h];
      e = (e > 0.f) ? e : NSLOPE * e;
      float w = __expf(e - fdec(m[(long)d * HEADS + h])) / den[(long)d * HEADS + h];
      const float* hp = hs + h * C;
      float* op = od + h * C;
#pragma unroll
      for (int c = lane; c < C; c += 32)
        atomicAdd(op + c, hp[c] * w);
    }
  }
}

// ---------- layer-1 finalize: bf16(elu(agg + b)) feeding GEMM2 ----------
__global__ void k_bias_elu_bf16(const float* __restrict__ a, const float* __restrict__ b,
                                __bf16* __restrict__ o, int colmask, long n) {
  long i = blockIdx.x * (long)blockDim.x + threadIdx.x;
  if (i >= n) return;
  float v = a[i] + b[i & colmask];
  v = (v > 0.f) ? v : (__expf(v) - 1.f);
  o[i] = (__bf16)v;
}

// ---------- layer-2 finalize: d_out = agg + b ----------
__global__ void k_bias_out(const float* __restrict__ a, const float* __restrict__ b,
                           float* __restrict__ o, int colmask, long n) {
  long i = blockIdx.x * (long)blockDim.x + threadIdx.x;
  if (i >= n) return;
  o[i] = a[i] + b[i & colmask];
}

// ------------------------------------------------------------------
extern "C" void kernel_launch(void* const* d_in, const int* in_sizes, int n_in,
                              void* d_out, int out_size, void* d_ws, size_t ws_size,
                              hipStream_t stream) {
  (void)in_sizes; (void)n_in; (void)out_size; (void)ws_size;
  const float* x   = (const float*)d_in[0];
  const int*   ei  = (const int*)  d_in[1];
  const float* W1  = (const float*)d_in[2];
  const float* as1 = (const float*)d_in[3];
  const float* ad1 = (const float*)d_in[4];
  const float* b1  = (const float*)d_in[5];
  const float* W2  = (const float*)d_in[6];
  const float* as2 = (const float*)d_in[7];
  const float* ad2 = (const float*)d_in[8];
  const float* b2  = (const float*)d_in[9];
  float* out = (float*)d_out;

  char* ws = (char*)d_ws;
  size_t off = 0;
  auto alloc = [&](size_t bytes) -> char* {
    off = (off + 255) & ~(size_t)255;
    char* p = ws + off; off += bytes; return p;
  };
  __bf16*   xbf  = (__bf16*)  alloc((size_t)NN * FIN * 2);
  __bf16*   w1t  = (__bf16*)  alloc((size_t)HC1 * FIN * 2);
  __bf16*   w2t  = (__bf16*)  alloc((size_t)C2C * HC1 * 2);
  float*    h1   = (float*)   alloc((size_t)NN * HC1 * 4);
  float*    ss1  = (float*)   alloc((size_t)NN * H1H * 4);
  float*    sd1  = (float*)   alloc((size_t)NN * H1H * 4);
  unsigned* m1   = (unsigned*)alloc((size_t)NN * H1H * 4);
  float*    den1 = (float*)   alloc((size_t)NN * H1H * 4);
  float*    agg1 = (float*)   alloc((size_t)NN * HC1 * 4);
  __bf16*   h1e  = (__bf16*)  alloc((size_t)NN * HC1 * 2);
  float*    h2   = (float*)   alloc((size_t)NN * C2C * 4);
  float*    ss2  = (float*)   alloc((size_t)NN * 4);
  float*    sd2  = (float*)   alloc((size_t)NN * 4);
  unsigned* m2   = (unsigned*)alloc((size_t)NN * 4);
  float*    den2 = (float*)   alloc((size_t)NN * 4);
  float*    agg2 = (float*)   alloc((size_t)NN * C2C * 4);

  const int B = 256;
  auto nb = [](long n, int b) { return (int)((n + b - 1) / b); };

  // --- precision prep ---
  k_cvt_bf16      <<<nb((long)NN * FIN, B), B, 0, stream>>>(x, xbf, (long)NN * FIN);
  k_transpose_bf16<<<nb((long)FIN * HC1, B), B, 0, stream>>>(W1, w1t, FIN, HC1);
  k_transpose_bf16<<<nb((long)HC1 * C2C, B), B, 0, stream>>>(W2, w2t, HC1, C2C);

  // --- layer 1: GEMM (10000x256 @ 256x1024) ---
  {
    long waves = (long)(NN / 16) * (HC1 / 64);          // 625 * 16 = 10000 waves
    k_gemm_bf16_wmma<<<nb(waves * 32, B), B, 0, stream>>>(xbf, w1t, h1, NN, HC1, FIN);
  }
  k_scores<H1H, C1C><<<nb((long)NN * H1H * 32, B), B, 0, stream>>>(h1, as1, ad1, ss1, sd1, NN);

  k_fill<<<nb((long)NN * H1H, B), B, 0, stream>>>(m1, 0u, (long)NN * H1H);
  k_fill<<<nb((long)NN * H1H, B), B, 0, stream>>>((unsigned*)den1, 0u, (long)NN * H1H);
  k_fill<<<nb((long)NN * HC1, B), B, 0, stream>>>((unsigned*)agg1, 0u, (long)NN * HC1);

  k_edge_max<H1H><<<nb((long)(EE + NN) * H1H, B), B, 0, stream>>>(ei, ss1, sd1, m1);
  k_edge_sum<H1H><<<nb((long)(EE + NN) * H1H, B), B, 0, stream>>>(ei, ss1, sd1, m1, den1);
  k_edge_agg<H1H, C1C><<<nb((long)(EE + NN) * 32, B), B, 0, stream>>>(ei, ss1, sd1, m1, den1, h1, agg1);
  k_bias_elu_bf16<<<nb((long)NN * HC1, B), B, 0, stream>>>(agg1, b1, h1e, HC1 - 1, (long)NN * HC1);

  // --- layer 2: GEMM (10000x1024 @ 1024x128) ---
  {
    long waves = (long)(NN / 16) * (C2C / 64);          // 625 * 2 = 1250 waves
    k_gemm_bf16_wmma<<<nb(waves * 32, B), B, 0, stream>>>(h1e, w2t, h2, NN, C2C, HC1);
  }
  k_scores<1, C2C><<<nb((long)NN * 32, B), B, 0, stream>>>(h2, as2, ad2, ss2, sd2, NN);

  k_fill<<<nb((long)NN, B), B, 0, stream>>>(m2, 0u, (long)NN);
  k_fill<<<nb((long)NN, B), B, 0, stream>>>((unsigned*)den2, 0u, (long)NN);
  k_fill<<<nb((long)NN * C2C, B), B, 0, stream>>>((unsigned*)agg2, 0u, (long)NN * C2C);

  k_edge_max<1><<<nb((long)(EE + NN), B), B, 0, stream>>>(ei, ss2, sd2, m2);
  k_edge_sum<1><<<nb((long)(EE + NN), B), B, 0, stream>>>(ei, ss2, sd2, m2, den2);
  k_edge_agg<1, C2C><<<nb((long)(EE + NN) * 32, B), B, 0, stream>>>(ei, ss2, sd2, m2, den2, h2, agg2);
  // mean over 1 head == identity; add bias into final output
  k_bias_out<<<nb((long)NN * C2C, B), B, 0, stream>>>(agg2, b2, out, C2C - 1, (long)NN * C2C);
}
